// MHA_29935922053585
// MI455X (gfx1250) — matrix operand order repaired and checked by reading, not compiled, not run on hardware
//
#include <hip/hip_runtime.h>

#define NH    16
#define AW    128
#define SEQ   2048
#define WDIM  2048
#define HD    2048   // NH*AW
#define BATCH 2
#define BK    64     // K-slab staged per TDM transfer

// LDS tile pitch after TDM padding: 64 elem * 2B = 128B data + 16B pad = 144B
#define BPITCH 144
#define BTILE  (128 * BPITCH)   // 18432 B per buffer

typedef __attribute__((ext_vector_type(16))) __bf16         v16bf;
typedef __attribute__((ext_vector_type(8)))  float          v8f;
typedef __attribute__((ext_vector_type(4)))  unsigned int   u32x4;
typedef __attribute__((ext_vector_type(4)))  float          f32x4;
typedef __attribute__((ext_vector_type(4)))  unsigned short u16x4;
typedef __attribute__((ext_vector_type(8)))  int            i32x8;
typedef __attribute__((ext_vector_type(4)))  int            i32x4;

union FragB { v16bf v; u32x4 q[2]; };

__device__ inline unsigned short f2bf(float f) {
  unsigned u = __builtin_bit_cast(unsigned, f);
  unsigned r = u + 0x7fffu + ((u >> 16) & 1u);   // round-to-nearest-even
  return (unsigned short)(r >> 16);
}

__device__ inline v8f v8f_zero() {
  v8f z = {0.f,0.f,0.f,0.f,0.f,0.f,0.f,0.f};
  return z;
}

// ---------------------------------------------------------------------------
// Tensor Data Mover: DMA a 128-row x BK-elem bf16 tile (row stride = K elems)
// from global into LDS at byte offset lds_off. D# fields per CDNA5 ISA §8.3-8.4.
// pad_enable inserts 4 DWORDs after every 32 DWORDs stored -> 144B LDS pitch
// (16B aligned for b128 LDS reads, odd bank stride -> conflict-free frags).
// ---------------------------------------------------------------------------
__device__ inline void tdm_load_tile(unsigned lds_off, const void* gptr, int K) {
  unsigned long long ga = (unsigned long long)(size_t)gptr;
  u32x4 g0;
  g0[0] = 1u;                                    // count=1 (user mode)
  g0[1] = lds_off;                               // D#.lds_addr (bytes)
  g0[2] = (unsigned)ga;                          // global_addr[31:0]
  g0[3] = (unsigned)((ga >> 32) & 0x01ffffffu)   // global_addr[56:32]
        | (2u << 30);                            // type = 2 ("image")
  i32x8 g1;
  g1[0] = (int)((1u << 16)                       // data_size = 2 bytes
        | (1u << 20)                             // pad_enable
        | (4u << 22)                             // pad_interval: 32 DWORDs
        | (3u << 25));                           // pad_amount:   4 DWORDs
  g1[1] = (int)((unsigned)K << 16);              // tensor_dim0[15:0]
  g1[2] = (int)(2048u << 16);                    // tensor_dim0[31:16]=0 | tensor_dim1[15:0]
  g1[3] = (int)((unsigned)BK << 16);             // tensor_dim1[31:16]=0 | tile_dim0 = BK
  g1[4] = 128;                                   // tile_dim1 = 128, tile_dim2 = 0
  g1[5] = K;                                     // tensor_dim0_stride[31:0]
  g1[6] = 0;                                     // stride0[47:32] | stride1[15:0]
  g1[7] = 0;
  i32x4 z4 = {0, 0, 0, 0};
#if __clang_major__ >= 23
  i32x8 z8 = {0, 0, 0, 0, 0, 0, 0, 0};
  __builtin_amdgcn_tensor_load_to_lds(g0, g1, z4, z4, z8, 0);
#else
  __builtin_amdgcn_tensor_load_to_lds(g0, g1, z4, z4, 0);
#endif
}

// ---------------------------------------------------------------------------
// f32 -> bf16 conversion, 4 elements per lane
// ---------------------------------------------------------------------------
__global__ __launch_bounds__(256) void cvt_f32_bf16(
    const float* __restrict__ in, unsigned short* __restrict__ out, int n4) {
  int i = blockIdx.x * 256 + threadIdx.x;
  if (i >= n4) return;
  f32x4 v = *reinterpret_cast<const f32x4*>(in + (size_t)i * 4);
  u16x4 o;
  o[0] = f2bf(v[0]); o[1] = f2bf(v[1]); o[2] = f2bf(v[2]); o[3] = f2bf(v[3]);
  *reinterpret_cast<u16x4*>(out + (size_t)i * 4) = o;
}

// ---------------------------------------------------------------------------
// GEMM  Y = A(bf16, MxK row-major) x W(bf16, NxK row-major)^T, f32 accum.
// Weight tiles staged to LDS via TDM (double buffered, s_wait_tensorcnt);
// A fragments stream direct from global (pipelined b128 clauses).
// mode 0: Q  -> rope + 1/sqrt(a) scale, store bf16 [b,h,n,a]
// mode 1: K  -> rope,                  store bf16 [b,h,n,a]
// mode 2: V  ->                        store bf16 [b,h,a,n] (transposed)
// mode 3: output projection            store f32  [b*n, HD]
// Block: 256 thr = 8 waves; block tile 128x128; wave tile 64(M)x32(N).
// ---------------------------------------------------------------------------
__global__ __launch_bounds__(256) void gemm_bf16(
    const unsigned short* __restrict__ A,
    const unsigned short* __restrict__ W0,
    const unsigned short* __restrict__ W1,
    const unsigned short* __restrict__ W2,
    unsigned short* __restrict__ O0,
    unsigned short* __restrict__ O1,
    unsigned short* __restrict__ O2,
    float* __restrict__ OF,
    int modeBase, int K)
{
  __shared__ __attribute__((aligned(128))) char smem[2 * BTILE];  // 36 KB
  const int mode = modeBase + blockIdx.z;
  const unsigned short* W = (blockIdx.z == 0) ? W0 : (blockIdx.z == 1) ? W1 : W2;
  const int lane  = threadIdx.x & 31;
  const int wave  = threadIdx.x >> 5;
  const int l15   = lane & 15;
  const int hi    = (lane >> 4) & 1;
  const int mBase = blockIdx.y * 128 + (wave & 1) * 64;
  const int nBase = blockIdx.x * 128;
  const int nLoc  = (wave >> 1) * 32;
  const unsigned smoff = (unsigned)(size_t)(void*)smem;   // LDS byte offset

  v8f acc[4][2];
  #pragma unroll
  for (int i = 0; i < 4; i++)
    #pragma unroll
    for (int j = 0; j < 2; j++) acc[i][j] = v8f_zero();

  // prologue: DMA first weight slab into buffer 0
  if (wave == 0)
    tdm_load_tile(smoff, W + (size_t)nBase * K, K);

  int cb = 0;
  for (int k = 0; k < K; k += BK) {
    if (wave == 0) {
      if (k + BK < K) {
        // issue next slab into the other buffer, then wait for current slab
        tdm_load_tile(smoff + (cb ? 0 : BTILE),
                      W + (size_t)nBase * K + k + BK, K);
        __builtin_amdgcn_s_wait_tensorcnt((short)1);  // TDM completes in order
      } else {
        __builtin_amdgcn_s_wait_tensorcnt((short)0);
      }
    }
    __syncthreads();   // publish current LDS slab to all waves

    const char* pB = smem + (cb ? BTILE : 0);
    #pragma unroll
    for (int kk = 0; kk < 2; kk++) {
      FragB a[4];
      #pragma unroll
      for (int mt = 0; mt < 4; mt++) {
        // A-matrix 16x32 layout: lanes 0-15 K 0..7/16..23, lanes 16-31 +8.
        const unsigned short* p =
            A + (size_t)(mBase + mt*16 + l15) * K + k + kk*32 + hi*8;
        a[mt].q[0] = *reinterpret_cast<const u32x4*>(p);
        a[mt].q[1] = *reinterpret_cast<const u32x4*>(p + 16);
      }
      FragB b[2];
      #pragma unroll
      for (int nt = 0; nt < 2; nt++) {
        // B-matrix 32x16: lane = N col; K 0..7/16..23 per lane (+8 for hi), from LDS.
        const char* p = pB + (size_t)(nLoc + nt*16 + l15) * BPITCH
                           + kk*64 + hi*16;
        b[nt].q[0] = *reinterpret_cast<const u32x4*>(p);
        b[nt].q[1] = *reinterpret_cast<const u32x4*>(p + 32);
      }
      #pragma unroll
      for (int mt = 0; mt < 4; mt++)
        #pragma unroll
        for (int nt = 0; nt < 2; nt++)
          acc[mt][nt] = __builtin_amdgcn_wmma_f32_16x16x32_bf16(
              false, a[mt].v, false, b[nt].v, (short)0, acc[mt][nt], false, false);
    }
    __syncthreads();   // all waves done reading before buffer is overwritten
    cb ^= 1;
  }

  const float ROPE_LN = 9.210340371976184f;   // ln(10000)
  #pragma unroll
  for (int mt = 0; mt < 4; mt++) {
    #pragma unroll
    for (int nt = 0; nt < 2; nt++) {
      int col = nBase + nLoc + nt*16 + l15;
      #pragma unroll
      for (int r = 0; r < 8; r++) {
        int   mg = mBase + mt*16 + r + hi*8;   // D layout row
        float v  = acc[mt][nt][r];
        if (mode == 3) {
          OF[(size_t)mg * HD + col] = v;
        } else {
          int b   = mg >> 11;
          int pos = mg & (SEQ - 1);
          int h   = col >> 7;
          int ai  = col & (AW - 1);
          if (mode <= 1) {
            // RoPE: interleaved pairs are adjacent columns = adjacent lanes.
            float fr = __expf(-(float)(ai >> 1) * (ROPE_LN / 63.f));
            float th = (float)pos * fr;
            float s, c; __sincosf(th, &s, &c);
            float partner = __shfl_xor(v, 1);
            v = (ai & 1) ? (v * c + partner * s) : (v * c - partner * s);
            if (mode == 0) v *= 0.08838834764831845f;   // 1/sqrt(128)
          }
          unsigned short bv = f2bf(v);
          if (mode == 2) {
            O2[((size_t)(b * NH + h) * AW + ai) * SEQ + pos] = bv;   // V^T
          } else {
            unsigned short* O = (mode == 0) ? O0 : O1;
            O[((size_t)(b * NH + h) * SEQ + pos) * AW + ai] = bv;
          }
        }
      }
    }
  }
}

// ---------------------------------------------------------------------------
// Causal flash attention. One wave = one 16-row query block; 32 keys/iter.
// Q,K in [b,h,n,a] bf16 (Q pre-scaled); V in [b,h,a,n] bf16.
// Output bf16 [b*n, HD] feeding the output projection.
// ---------------------------------------------------------------------------
__global__ __launch_bounds__(128) void flash_attn(
    const unsigned short* __restrict__ Qb,
    const unsigned short* __restrict__ Kb,
    const unsigned short* __restrict__ Vtb,
    unsigned short* __restrict__ AttnB)
{
  __shared__ unsigned short lds[4 * 16 * 32];     // 1KB per wave P-tile
  const int lane = threadIdx.x & 31;
  const int wave = threadIdx.x >> 5;
  const int l15  = lane & 15;
  const int hi   = (lane >> 4) & 1;
  const int qb   = blockIdx.x * 4 + wave;
  const int bh   = blockIdx.y;
  unsigned short* pl = lds + wave * (16 * 32);

  // Q fragments: 16 rows x 128, kept in registers (4 A-frags).
  FragB qf[4];
  const unsigned short* qrow = Qb + ((size_t)bh * SEQ + qb*16 + l15) * AW;
  #pragma unroll
  for (int kk = 0; kk < 4; kk++) {
    const unsigned short* p = qrow + kk*32 + hi*8;
    qf[kk].q[0] = *reinterpret_cast<const u32x4*>(p);
    qf[kk].q[1] = *reinterpret_cast<const u32x4*>(p + 16);
  }

  v8f o[8];
  #pragma unroll
  for (int i = 0; i < 8; i++) o[i] = v8f_zero();
  float mrow[8], lrow[8];
  #pragma unroll
  for (int r = 0; r < 8; r++) { mrow[r] = -3.0e38f; lrow[r] = 0.f; }

  const int qTop = qb * 16;
  for (int kbase = 0; kbase <= qTop; kbase += 32) {
    // S = Q x K^T for 32 keys (two 16x16 tiles)
    v8f s0 = v8f_zero(), s1 = v8f_zero();
    #pragma unroll
    for (int kk = 0; kk < 4; kk++) {
      FragB k0, k1;
      const unsigned short* p0 =
          Kb + ((size_t)bh * SEQ + kbase + l15) * AW + kk*32 + hi*8;
      k0.q[0] = *reinterpret_cast<const u32x4*>(p0);
      k0.q[1] = *reinterpret_cast<const u32x4*>(p0 + 16);
      const unsigned short* p1 = p0 + 16 * AW;
      k1.q[0] = *reinterpret_cast<const u32x4*>(p1);
      k1.q[1] = *reinterpret_cast<const u32x4*>(p1 + 16);
      s0 = __builtin_amdgcn_wmma_f32_16x16x32_bf16(false, qf[kk].v, false, k0.v,
                                                   (short)0, s0, false, false);
      s1 = __builtin_amdgcn_wmma_f32_16x16x32_bf16(false, qf[kk].v, false, k1.v,
                                                   (short)0, s1, false, false);
    }
    // causal mask (only the diagonal-overlapping iteration needs it)
    if (kbase + 31 > qTop) {
      int key0 = kbase + l15, key1 = key0 + 16;
      #pragma unroll
      for (int r = 0; r < 8; r++) {
        int qr = qTop + r + hi*8;
        if (key0 > qr) s0[r] = -3.0e38f;
        if (key1 > qr) s1[r] = -3.0e38f;
      }
    }
    // online softmax; row values live across 16-lane halves -> xor shuffles
    #pragma unroll
    for (int r = 0; r < 8; r++) {
      float mx = fmaxf(s0[r], s1[r]);
      mx = fmaxf(mx, __shfl_xor(mx, 1));
      mx = fmaxf(mx, __shfl_xor(mx, 2));
      mx = fmaxf(mx, __shfl_xor(mx, 4));
      mx = fmaxf(mx, __shfl_xor(mx, 8));
      float mnew  = fmaxf(mrow[r], mx);
      float alpha = __expf(mrow[r] - mnew);
      mrow[r] = mnew;
      float p0 = __expf(s0[r] - mnew);
      float p1 = __expf(s1[r] - mnew);
      float rs = p0 + p1;
      rs += __shfl_xor(rs, 1);
      rs += __shfl_xor(rs, 2);
      rs += __shfl_xor(rs, 4);
      rs += __shfl_xor(rs, 8);
      lrow[r] = lrow[r] * alpha + rs;
      #pragma unroll
      for (int na = 0; na < 8; na++) o[na][r] *= alpha;
      // stage P (D layout) into LDS, row-major 16x32 bf16
      int m = r + hi*8;
      pl[m*32 + l15]      = f2bf(p0);
      pl[m*32 + 16 + l15] = f2bf(p1);
    }
    // reload P in A-matrix 16x32 layout (same-wave LDS ops are in-order)
    FragB pf;
    {
      const unsigned short* p = pl + l15*32 + hi*8;
      pf.q[0] = *reinterpret_cast<const u32x4*>(p);
      pf.q[1] = *reinterpret_cast<const u32x4*>(p + 16);
    }
    // O += P x V  (V pre-transposed -> contiguous B-frag loads)
    #pragma unroll
    for (int na = 0; na < 8; na++) {
      FragB vf;
      const unsigned short* p =
          Vtb + ((size_t)bh * AW + na*16 + l15) * SEQ + kbase + hi*8;
      vf.q[0] = *reinterpret_cast<const u32x4*>(p);
      vf.q[1] = *reinterpret_cast<const u32x4*>(p + 16);
      o[na] = __builtin_amdgcn_wmma_f32_16x16x32_bf16(false, pf.v, false, vf.v,
                                                      (short)0, o[na], false, false);
    }
  }

  // normalize + store bf16 to [b*n, HD]
  const int b = bh >> 4, h = bh & 15;
  #pragma unroll
  for (int r = 0; r < 8; r++) {
    float inv = 1.f / lrow[r];
    int m = qb*16 + r + hi*8;
    size_t rowOff = ((size_t)b * SEQ + m) * HD + (size_t)h * AW;
    #pragma unroll
    for (int na = 0; na < 8; na++)
      AttnB[rowOff + na*16 + l15] = f2bf(o[na][r] * inv);
  }
}

// ---------------------------------------------------------------------------
extern "C" void kernel_launch(void* const* d_in, const int* in_sizes, int n_in,
                              void* d_out, int out_size, void* d_ws, size_t ws_size,
                              hipStream_t stream) {
  (void)in_sizes; (void)n_in; (void)out_size; (void)ws_size;
  const float* q  = (const float*)d_in[0];
  const float* Wq = (const float*)d_in[1];
  const float* Wk = (const float*)d_in[2];
  const float* Wv = (const float*)d_in[3];
  const float* Wo = (const float*)d_in[4];
  float* out = (float*)d_out;

  const size_t SZ_Q = (size_t)BATCH * SEQ * WDIM;   // 8,388,608 elems
  const size_t SZ_W = (size_t)HD * WDIM;            // 4,194,304 elems

  char* ws = (char*)d_ws;
  unsigned short* qb    = (unsigned short*)(ws);                       // 16 MB
  unsigned short* wqb   = (unsigned short*)(ws + 16777216);            //  8 MB
  unsigned short* wkb   = (unsigned short*)(ws + 25165824);            //  8 MB
  unsigned short* wvb   = (unsigned short*)(ws + 33554432);            //  8 MB
  unsigned short* wob   = (unsigned short*)(ws + 41943040);            //  8 MB
  unsigned short* Qb    = (unsigned short*)(ws + 50331648);            // 16 MB
  unsigned short* Kb    = (unsigned short*)(ws + 67108864);            // 16 MB
  unsigned short* Vtb   = (unsigned short*)(ws + 83886080);            // 16 MB
  unsigned short* AttnB = (unsigned short*)(ws + 100663296);           // 16 MB

  // 1) downconvert activations + weights to bf16
  cvt_f32_bf16<<<dim3((unsigned)(SZ_Q/4/256)), 256, 0, stream>>>(q,  qb,  (int)(SZ_Q/4));
  cvt_f32_bf16<<<dim3((unsigned)(SZ_W/4/256)), 256, 0, stream>>>(Wq, wqb, (int)(SZ_W/4));
  cvt_f32_bf16<<<dim3((unsigned)(SZ_W/4/256)), 256, 0, stream>>>(Wk, wkb, (int)(SZ_W/4));
  cvt_f32_bf16<<<dim3((unsigned)(SZ_W/4/256)), 256, 0, stream>>>(Wv, wvb, (int)(SZ_W/4));
  cvt_f32_bf16<<<dim3((unsigned)(SZ_W/4/256)), 256, 0, stream>>>(Wo, wob, (int)(SZ_W/4));

  // 2) fused QKV projection + RoPE + head-split (+ V transpose)
  gemm_bf16<<<dim3(HD/128, (BATCH*SEQ)/128, 3), 256, 0, stream>>>(
      qb, wqb, wkb, wvb, Qb, Kb, Vtb, nullptr, 0, WDIM);

  // 3) causal flash attention
  flash_attn<<<dim3((SEQ/16)/4, BATCH*NH), 128, 0, stream>>>(Qb, Kb, Vtb, AttnB);

  // 4) output projection -> f32
  gemm_bf16<<<dim3(WDIM/128, (BATCH*SEQ)/128, 1), 256, 0, stream>>>(
      AttnB, wob, nullptr, nullptr, nullptr, nullptr, nullptr, out, 3, HD);
}
